// LSHmodule_40578851013249
// MI455X (gfx1250) — matrix operand order repaired and checked by reading, not compiled, not run on hardware
//
#include <hip/hip_runtime.h>
#include <math.h>

// ---------------------------------------------------------------------------
// Problem constants (B=2, S=2048, D=512, 64 buckets)
// ---------------------------------------------------------------------------
#define BATCH 2
#define SEQ   2048
#define DIM   512
#define NBUCK 64
#define NROWS (BATCH * SEQ)          // 4096 token rows
#define SCALE 0.044194173824159216f  // 1/sqrt(512)
#define TQ    32                     // query rows per attention block

// ---------------------------------------------------------------------------
// bf16 helpers (RNE)
// ---------------------------------------------------------------------------
__device__ __forceinline__ unsigned short f2bf(float f) {
  unsigned int u = __float_as_uint(f);
  u += 0x7FFFu + ((u >> 16) & 1u);
  return (unsigned short)(u >> 16);
}
__device__ __forceinline__ float bf2f(unsigned short h) {
  return __uint_as_float(((unsigned int)h) << 16);
}

// ---------------------------------------------------------------------------
// WMMA fragment plumbing (CDNA5 wave32, 16x16x32 bf16 -> f32)
//   A (16xK): lane M = lane&15 ; dword p holds K = 2*p + 16*half
//   B (Kx16): lane N = lane&15 ; dword p holds K = 2*p + 16*half
//   C/D     : lane N = lane&15 ; element r is row M = r + 8*half
// ---------------------------------------------------------------------------
typedef __attribute__((ext_vector_type(16))) __bf16 v16bf;
typedef __attribute__((ext_vector_type(8)))  float  v8f;

union Frag {
  v16bf v;
  unsigned int u[8];
};

__device__ __forceinline__ int ka_idx(int p, int half) {
  return 2 * p + 16 * half;
}
__device__ __forceinline__ int kb_idx(int p, int half) {
  return 2 * p + 16 * half;
}
__device__ __forceinline__ v8f wmma_bf16(const Frag& a, const Frag& b, v8f c) {
  return __builtin_amdgcn_wmma_f32_16x16x32_bf16(false, a.v, false, b.v,
                                                 (short)0, c, false, false);
}

// ---------------------------------------------------------------------------
// Kernel 1: f32 -> (bf16 hi, bf16 lo) split (for bf16x3 fp32-accurate GEMMs)
// ---------------------------------------------------------------------------
__global__ void cvt_hilo_kernel(const float* __restrict__ src,
                                unsigned short* __restrict__ h,
                                unsigned short* __restrict__ l, int n) {
  int i = blockIdx.x * blockDim.x + threadIdx.x;
  if (i < n) {
    float f = src[i];
    unsigned short hi = f2bf(f);
    h[i] = hi;
    l[i] = f2bf(f - bf2f(hi));
  }
}

// ---------------------------------------------------------------------------
// Kernel 2: projections. q = x@Wq^T + bq (f32 out), v = x@Wv^T + bv (bf16,
// stored TRANSPOSED as vT[b][d][s] so P@V B-fragments are contiguous dwords).
// bf16x3: acc += xl*Wh + xh*Wl + xh*Wh. One 16x16 tile per wave, K=512.
// grid (NROWS/16, DIM/64, 2[q,v]), block 128 (4 waves).
// ---------------------------------------------------------------------------
__global__ void __launch_bounds__(128) proj_kernel(
    const unsigned short* __restrict__ xh, const unsigned short* __restrict__ xl,
    const unsigned short* __restrict__ wqh, const unsigned short* __restrict__ wql,
    const unsigned short* __restrict__ wvh, const unsigned short* __restrict__ wvl,
    const float* __restrict__ bq, const float* __restrict__ bv,
    float* __restrict__ qf, unsigned short* __restrict__ vT) {
  const int wave = threadIdx.x >> 5, lane = threadIdx.x & 31;
  const int half = lane >> 4, nm = lane & 15;
  const int t0 = blockIdx.x * 16;
  const int c0 = (blockIdx.y * 4 + wave) * 16;
  const bool isV = (blockIdx.z == 1);
  const unsigned short* wh = isV ? wvh : wqh;
  const unsigned short* wl = isV ? wvl : wql;
  const float* bias = isV ? bv : bq;

  v8f acc = {0.f, 0.f, 0.f, 0.f, 0.f, 0.f, 0.f, 0.f};
  for (int k0 = 0; k0 < DIM; k0 += 32) {
    Frag ah, al, bh, bl;
#pragma unroll
    for (int p = 0; p < 8; ++p) {
      const size_t aoff = (size_t)(t0 + nm) * DIM + k0 + ka_idx(p, half);
      ah.u[p] = *(const unsigned int*)(xh + aoff);
      al.u[p] = *(const unsigned int*)(xl + aoff);
      const size_t boff = (size_t)(c0 + nm) * DIM + k0 + kb_idx(p, half);
      bh.u[p] = *(const unsigned int*)(wh + boff);
      bl.u[p] = *(const unsigned int*)(wl + boff);
    }
    acc = wmma_bf16(al, bh, acc);
    acc = wmma_bf16(ah, bl, acc);
    acc = wmma_bf16(ah, bh, acc);
  }
  const float bs = bias[c0 + nm];
#pragma unroll
  for (int r = 0; r < 8; ++r) {
    const int tg = t0 + r + 8 * half;   // global token row
    const float val = acc[r] + bs;
    if (!isV) {
      qf[(size_t)tg * DIM + c0 + nm] = val;
    } else {
      const int b = tg >> 11, s = tg & (SEQ - 1);
      vT[((size_t)(b * DIM + c0 + nm)) * SEQ + s] = f2bf(val);
    }
  }
}

// ---------------------------------------------------------------------------
// Kernel 3: per-token prep — q hi/lo split + LSH bucket id. One wave / token.
// ---------------------------------------------------------------------------
__global__ void __launch_bounds__(32) prep_kernel(
    const float* __restrict__ qf, const float* __restrict__ H,
    unsigned short* __restrict__ qh, unsigned short* __restrict__ ql,
    int* __restrict__ bkt) {
  const int tok = blockIdx.x;
  const int lane = threadIdx.x;
  const float* qrow = qf + (size_t)tok * DIM;
  float pp[6] = {0.f, 0.f, 0.f, 0.f, 0.f, 0.f};
#pragma unroll
  for (int j = 0; j < 16; ++j) {
    const int d = lane + 32 * j;
    const float qv = qrow[d];
    const unsigned short hi = f2bf(qv);
    qh[(size_t)tok * DIM + d] = hi;
    ql[(size_t)tok * DIM + d] = f2bf(qv - bf2f(hi));
    const float* Hr = H + d * 6;
#pragma unroll
    for (int nn = 0; nn < 6; ++nn) pp[nn] += qv * Hr[nn];
  }
#pragma unroll
  for (int off = 16; off > 0; off >>= 1) {
#pragma unroll
    for (int nn = 0; nn < 6; ++nn) pp[nn] += __shfl_xor(pp[nn], off, 32);
  }
  if (lane == 0) {
    int bb = 0;
#pragma unroll
    for (int nn = 0; nn < 6; ++nn) {
      if (pp[nn] + H[DIM * 6 + nn] >= 0.f) bb |= (1 << nn);  // +bias row (const-1)
    }
    bkt[tok] = bb;
  }
}

// ---------------------------------------------------------------------------
// Kernel 4: S_full[b][t][u] = scale * (q_t . q_u), bf16x3 accurate.
// One 16x16 tile per wave. grid (SEQ/16, SEQ/64, BATCH), block 128.
// ---------------------------------------------------------------------------
__global__ void __launch_bounds__(128) scores_kernel(
    const unsigned short* __restrict__ qh, const unsigned short* __restrict__ ql,
    float* __restrict__ S) {
  const int wave = threadIdx.x >> 5, lane = threadIdx.x & 31;
  const int half = lane >> 4, nm = lane & 15;
  const int b = blockIdx.z;
  const int t0 = blockIdx.x * 16;
  const int u0 = (blockIdx.y * 4 + wave) * 16;
  const size_t qbase = (size_t)b * SEQ * DIM;

  v8f acc = {0.f, 0.f, 0.f, 0.f, 0.f, 0.f, 0.f, 0.f};
  for (int k0 = 0; k0 < DIM; k0 += 32) {
    Frag ah, al, bh, bl;
#pragma unroll
    for (int p = 0; p < 8; ++p) {
      const size_t aoff = qbase + (size_t)(t0 + nm) * DIM + k0 + ka_idx(p, half);
      ah.u[p] = *(const unsigned int*)(qh + aoff);
      al.u[p] = *(const unsigned int*)(ql + aoff);
      const size_t boff = qbase + (size_t)(u0 + nm) * DIM + k0 + kb_idx(p, half);
      bh.u[p] = *(const unsigned int*)(qh + boff);
      bl.u[p] = *(const unsigned int*)(ql + boff);
    }
    acc = wmma_bf16(al, bh, acc);
    acc = wmma_bf16(ah, bl, acc);
    acc = wmma_bf16(ah, bh, acc);
  }
  float* Sb = S + (size_t)b * SEQ * SEQ;
#pragma unroll
  for (int r = 0; r < 8; ++r)
    Sb[(size_t)(t0 + r + 8 * half) * SEQ + u0 + nm] = acc[r] * SCALE;
}

// ---------------------------------------------------------------------------
// Kernel 5: fused masked-softmax + P@V over all 64 buckets.
// Block = 256 threads (8 waves) owns one (batch, 32-query-row tile); loops
// buckets with persistent f32 accumulators (32x512 out: 8 waves x 64 cols,
// each wave does 2 M-tiles so every V B-fragment feeds 2 WMMAs -> halves L2
// traffic on V). Per bucket: logit l = (rb==i || bu==i) ? 0 : s ; max-
// subtracted exp; masked columns store p=0 (numerator) but count exp(0-m) in
// the denominator. P tile in LDS: 32 x (2048+2) bf16 = 128.1 KB (<320 KB WGP).
// ---------------------------------------------------------------------------
#define PSTRIDE_W 1025  // dwords per P row: (2048+2) bf16 / 2

__global__ void __launch_bounds__(256) attn_kernel(
    const float* __restrict__ S, const unsigned short* __restrict__ vT,
    const int* __restrict__ bkt, float* __restrict__ out) {
  extern __shared__ char smem[];
  unsigned int* p_ldsw = (unsigned int*)smem;                      // 32*1025 dwords
  float* red  = (float*)(smem + TQ * PSTRIDE_W * 4);               // 32*8
  float* mrow = red + 256;                                         // 32
  float* dinv = mrow + TQ;                                         // 32
  unsigned char* bktl = (unsigned char*)(dinv + TQ);               // 2048

  const int b  = blockIdx.y;
  const int t0 = blockIdx.x * TQ;
  const int tid = threadIdx.x;
  const int wave = tid >> 5, lane = tid & 31;
  const int half = lane >> 4, nm = lane & 15;
  const int r = tid >> 3, sub = tid & 7;   // 32 rows x 8 threads/row

  for (int u = tid; u < SEQ; u += 256) bktl[u] = (unsigned char)bkt[b * SEQ + u];
  __syncthreads();
  const int rb = bktl[t0 + r];

  const float* srow = S + ((size_t)(b * SEQ + t0 + r)) * SEQ;
  const unsigned short* vTb = vT + (size_t)b * DIM * SEQ;

  const v8f vzero = {0.f, 0.f, 0.f, 0.f, 0.f, 0.f, 0.f, 0.f};
  v8f accp[2][4];
#pragma unroll
  for (int mt = 0; mt < 2; ++mt)
#pragma unroll
    for (int nt = 0; nt < 4; ++nt) accp[mt][nt] = vzero;

  for (int i = 0; i < NBUCK; ++i) {
    // ---- pass 1: row max over masked logits (8 threads per row) ----
    float m = -3.0e38f;
    for (int k = 0; k < 128; ++k) {
      const int u0 = 2 * sub + 16 * k;
      const float2 sv = *(const float2*)(srow + u0);
      const float l0 = ((rb == i) | (bktl[u0] == i)) ? 0.f : sv.x;
      const float l1 = ((rb == i) | (bktl[u0 + 1] == i)) ? 0.f : sv.y;
      m = fmaxf(m, fmaxf(l0, l1));
    }
    red[r * 8 + sub] = m;
    __syncthreads();
    if (sub == 0) {
      float mm = red[r * 8];
      for (int j = 1; j < 8; ++j) mm = fmaxf(mm, red[r * 8 + j]);
      mrow[r] = mm;
    }
    __syncthreads();
    const float mr = mrow[r];

    // ---- pass 2: p = exp(l-m) -> LDS (bf16), denominator sum ----
    float dsum = 0.f;
    for (int k = 0; k < 128; ++k) {
      const int u0 = 2 * sub + 16 * k;
      const float2 sv = *(const float2*)(srow + u0);
      const bool m0 = (bktl[u0] == i), m1 = (bktl[u0 + 1] == i);
      const float l0 = ((rb == i) | m0) ? 0.f : sv.x;
      const float l1 = ((rb == i) | m1) ? 0.f : sv.y;
      const float e0 = __expf(l0 - mr);
      const float e1 = __expf(l1 - mr);
      dsum += e0 + e1;
      const unsigned int pk = (unsigned int)f2bf(m0 ? 0.f : e0) |
                              ((unsigned int)f2bf(m1 ? 0.f : e1) << 16);
      p_ldsw[r * PSTRIDE_W + sub + 8 * k] = pk;
    }
    red[r * 8 + sub] = dsum;
    __syncthreads();
    if (sub == 0) {
      float dd = 0.f;
      for (int j = 0; j < 8; ++j) dd += red[r * 8 + j];
      dinv[r] = 1.0f / dd;
    }
    __syncthreads();

    // ---- phase 2: out += (P @ V) * dinv ; wave = 64 cols x 2 M-tiles ----
    v8f lacc[2][4];
#pragma unroll
    for (int mt = 0; mt < 2; ++mt)
#pragma unroll
      for (int nt = 0; nt < 4; ++nt) lacc[mt][nt] = vzero;

    for (int k0 = 0; k0 < SEQ; k0 += 32) {
      Frag a0, a1;
#pragma unroll
      for (int p = 0; p < 8; ++p) {
        const int kw = (k0 + ka_idx(p, half)) >> 1;
        a0.u[p] = p_ldsw[nm * PSTRIDE_W + kw];
        a1.u[p] = p_ldsw[(16 + nm) * PSTRIDE_W + kw];
      }
#pragma unroll
      for (int nt = 0; nt < 4; ++nt) {
        const int dcol = wave * 64 + nt * 16 + nm;
        Frag bf_;
#pragma unroll
        for (int p = 0; p < 8; ++p)
          bf_.u[p] = *(const unsigned int*)(vTb + (size_t)dcol * SEQ + k0 +
                                            kb_idx(p, half));
        lacc[0][nt] = wmma_bf16(a0, bf_, lacc[0][nt]);
        lacc[1][nt] = wmma_bf16(a1, bf_, lacc[1][nt]);
      }
    }
    float dv[2][8];
#pragma unroll
    for (int mt = 0; mt < 2; ++mt)
#pragma unroll
      for (int rr = 0; rr < 8; ++rr) dv[mt][rr] = dinv[mt * 16 + rr + 8 * half];
#pragma unroll
    for (int mt = 0; mt < 2; ++mt)
#pragma unroll
      for (int nt = 0; nt < 4; ++nt)
#pragma unroll
        for (int rr = 0; rr < 8; ++rr)
          accp[mt][nt][rr] += lacc[mt][nt][rr] * dv[mt][rr];
    __syncthreads();  // protect p_lds before next bucket overwrites it
  }

  // ---- final store (each block owns its 32x512 tile exclusively) ----
#pragma unroll
  for (int mt = 0; mt < 2; ++mt)
#pragma unroll
    for (int nt = 0; nt < 4; ++nt) {
      const int dcol = wave * 64 + nt * 16 + nm;
#pragma unroll
      for (int rr = 0; rr < 8; ++rr)
        out[(size_t)(b * SEQ + t0 + mt * 16 + rr + 8 * half) * DIM + dcol] =
            accp[mt][nt][rr];
    }
}

// ---------------------------------------------------------------------------
// Host launcher
// ---------------------------------------------------------------------------
extern "C" void kernel_launch(void* const* d_in, const int* in_sizes, int n_in,
                              void* d_out, int out_size, void* d_ws, size_t ws_size,
                              hipStream_t stream) {
  (void)in_sizes; (void)n_in; (void)out_size; (void)ws_size;
  const float* x  = (const float*)d_in[0];
  const float* Wq = (const float*)d_in[1];
  const float* bq = (const float*)d_in[2];
  const float* Wv = (const float*)d_in[3];
  const float* bv = (const float*)d_in[4];
  const float* H  = (const float*)d_in[5];
  float* out = (float*)d_out;

  char* ws = (char*)d_ws;
  size_t off = 0;
  auto alloc = [&](size_t bytes) -> void* {
    void* p = ws + off;
    off += (bytes + 255) & ~(size_t)255;
    return p;
  };
  unsigned short* xh  = (unsigned short*)alloc((size_t)NROWS * DIM * 2);
  unsigned short* xl  = (unsigned short*)alloc((size_t)NROWS * DIM * 2);
  unsigned short* wqh = (unsigned short*)alloc((size_t)DIM * DIM * 2);
  unsigned short* wql = (unsigned short*)alloc((size_t)DIM * DIM * 2);
  unsigned short* wvh = (unsigned short*)alloc((size_t)DIM * DIM * 2);
  unsigned short* wvl = (unsigned short*)alloc((size_t)DIM * DIM * 2);
  float*          qf  = (float*)alloc((size_t)NROWS * DIM * 4);
  unsigned short* qh  = (unsigned short*)alloc((size_t)NROWS * DIM * 2);
  unsigned short* ql  = (unsigned short*)alloc((size_t)NROWS * DIM * 2);
  unsigned short* vT  = (unsigned short*)alloc((size_t)BATCH * DIM * SEQ * 2);
  int*            bkt = (int*)alloc((size_t)NROWS * 4);
  float*          S   = (float*)alloc((size_t)BATCH * SEQ * SEQ * 4);

  // 1) hi/lo splits of x, Wq, Wv
  {
    const int nx = NROWS * DIM, nw = DIM * DIM;
    cvt_hilo_kernel<<<(nx + 255) / 256, 256, 0, stream>>>(x, xh, xl, nx);
    cvt_hilo_kernel<<<(nw + 255) / 256, 256, 0, stream>>>(Wq, wqh, wql, nw);
    cvt_hilo_kernel<<<(nw + 255) / 256, 256, 0, stream>>>(Wv, wvh, wvl, nw);
  }
  // 2) projections: q (f32), vT (bf16, transposed)
  proj_kernel<<<dim3(NROWS / 16, DIM / 64, 2), 128, 0, stream>>>(
      xh, xl, wqh, wql, wvh, wvl, bq, bv, qf, vT);
  // 3) q hi/lo + bucket ids
  prep_kernel<<<NROWS, 32, 0, stream>>>(qf, H, qh, ql, bkt);
  // 4) full score matrix, scaled
  scores_kernel<<<dim3(SEQ / 16, SEQ / 64, BATCH), 128, 0, stream>>>(qh, ql, S);
  // 5) fused 64-bucket masked softmax-attention
  const int shmem = TQ * PSTRIDE_W * 4 + 256 * 4 + TQ * 4 + TQ * 4 + SEQ;
  hipFuncSetAttribute(reinterpret_cast<const void*>(attn_kernel),
                      hipFuncAttributeMaxDynamicSharedMemorySize, shmem);
  attn_kernel<<<dim3(SEQ / TQ, BATCH), 256, shmem, stream>>>(S, vT, bkt, out);
}
